// SUPRARecurrentFixedGamma_10024453668954
// MI455X (gfx1250) — matrix operand order repaired and checked by reading, not compiled, not run on hardware
//
#include <hip/hip_runtime.h>
#include <hip/hip_bf16.h>

typedef __bf16 bf16;
typedef __attribute__((ext_vector_type(16))) __bf16 v16bf;
typedef __attribute__((ext_vector_type(8)))  __bf16 v8bf;
typedef __attribute__((ext_vector_type(8)))  float  v8f;

#define D_DIM 1024
#define S_DIM 2048
#define B_DIM 4
#define CHUNK 64
#define SLICE 64
#define NCHUNK (S_DIM / CHUNK)
#define NSLICE (D_DIM / SLICE)
#define ST_STRIDE 1028   /* f32 elements per state row (padded) */
#define A_STRIDE  68     /* f32 elements per A row (padded)     */
#define VT_STRIDE 80     /* bf16 elements per Vt row (32B-aligned rows) */
#define KT_STRIDE 80     /* bf16 elements per Kt row */
#define EPS_V 1e-6f

// ---------------------------------------------------------------- WMMA helpers

__device__ __forceinline__ v8f wmma_bf(v16bf a, v16bf b, v8f c) {
  return __builtin_amdgcn_wmma_f32_16x16x32_bf16(false, a, false, b, (short)0, c,
                                                 false, false);
}

__device__ __forceinline__ v8f zero8() {
  v8f r = {0.f, 0.f, 0.f, 0.f, 0.f, 0.f, 0.f, 0.f};
  return r;
}

__device__ __forceinline__ v16bf combine8(v8bf lo, v8bf hi) {
  v16bf r;
#pragma unroll
  for (int i = 0; i < 8; ++i) { r[i] = lo[i]; r[i + 8] = hi[i]; }
  return r;
}

// A fragment (16x32 bf16). Lane holds row m=(lane&15); lanes 0-15 carry
// K = k0+0..15, lanes 16-31 carry K = k0+16..31.
__device__ __forceinline__ v16bf loadA_bf(const bf16* row, int k0, int lane) {
  int ko = k0 + ((lane >> 4) << 4);
  v8bf lo = *(const v8bf*)(row + ko);
  v8bf hi = *(const v8bf*)(row + ko + 8);
  return combine8(lo, hi);
}
__device__ __forceinline__ v16bf loadA_f32(const float* row, int k0, int lane) {
  int ko = k0 + ((lane >> 4) << 4);
  v16bf r;
#pragma unroll
  for (int i = 0; i < 8; ++i) {
    r[i]     = (bf16)row[ko + i];
    r[i + 8] = (bf16)row[ko + 8 + i];
  }
  return r;
}
// B fragment (32x16 bf16). Lane holds column n=(lane&15); 16 contiguous K at
// k0 + (lane>=16 ? 16 : 0). `colrow` points at the column's contiguous storage.
__device__ __forceinline__ v16bf loadB_bf(const bf16* colrow, int k0, int lane) {
  int ko = k0 + ((lane >> 4) << 4);
  return *(const v16bf*)(colrow + ko);
}
__device__ __forceinline__ v16bf loadB_f32(const float* colrow, int k0, int lane) {
  int ko = k0 + ((lane >> 4) << 4);
  v16bf r;
#pragma unroll
  for (int i = 0; i < 16; ++i) r[i] = (bf16)colrow[ko + i];
  return r;
}

// ---------------------------------------------------------------- f32 -> bf16

__global__ void cvt_kernel(const float* __restrict__ in, bf16* __restrict__ out,
                           int n) {
  int i = (blockIdx.x * blockDim.x + threadIdx.x) * 8;
  if (i >= n) return;
  v8bf o;
#pragma unroll
  for (int k = 0; k < 8; ++k) o[k] = (bf16)in[i + k];
  *(v8bf*)(out + i) = o;
}

// ---------------------------------------------------------------- projections
// Y = relu?(Xbf @ Wbf^T + b), written as bf16. Block = 8 waves; each wave owns a
// 32x32 output tile (2x2 WMMA tiles). Block tile 128(M) x 64(N). grid.z selects
// q/k/v.

__global__ __launch_bounds__(256) void proj_kernel(
    const bf16* __restrict__ X, const bf16* __restrict__ Wq,
    const bf16* __restrict__ Wk, const bf16* __restrict__ Wv,
    const float* __restrict__ bq, const float* __restrict__ bk,
    const float* __restrict__ bv, bf16* __restrict__ Qo, bf16* __restrict__ Ko,
    bf16* __restrict__ Vo) {
  const int which = blockIdx.z;
  const bf16* W = (which == 0) ? Wq : (which == 1) ? Wk : Wv;
  const float* bias = (which == 0) ? bq : (which == 1) ? bk : bv;
  bf16* O = (which == 0) ? Qo : (which == 1) ? Ko : Vo;
  const bool do_relu = (which < 2);

  const int tid = threadIdx.x, lane = tid & 31, w = tid >> 5;
  const int mBase = blockIdx.x * 128 + (w >> 1) * 32;
  const int nBase = blockIdx.y * 64 + (w & 1) * 32;

  const bf16* a0row = X + (size_t)(mBase + (lane & 15)) * D_DIM;
  const bf16* a1row = a0row + 16 * D_DIM;
  const bf16* b0row = W + (size_t)(nBase + (lane & 15)) * D_DIM;
  const bf16* b1row = b0row + 16 * D_DIM;

  v8f acc00 = zero8(), acc01 = zero8(), acc10 = zero8(), acc11 = zero8();

#pragma unroll 4
  for (int kk = 0; kk < D_DIM / 32; ++kk) {
    const int k0 = kk * 32;
    v16bf a0 = loadA_bf(a0row, k0, lane);
    v16bf a1 = loadA_bf(a1row, k0, lane);
    v16bf b0 = loadB_bf(b0row, k0, lane);
    v16bf b1 = loadB_bf(b1row, k0, lane);
    acc00 = wmma_bf(a0, b0, acc00);
    acc01 = wmma_bf(a0, b1, acc01);
    acc10 = wmma_bf(a1, b0, acc10);
    acc11 = wmma_bf(a1, b1, acc11);
  }

  const int nloc = lane & 15, mlo = (lane >> 4) * 8;
#pragma unroll
  for (int j = 0; j < 2; ++j) {
    const int col = nBase + j * 16 + nloc;
    const float bc = bias[col];
#pragma unroll
    for (int i = 0; i < 2; ++i) {
      v8f acc = (i == 0) ? (j == 0 ? acc00 : acc01) : (j == 0 ? acc10 : acc11);
#pragma unroll
      for (int r = 0; r < 8; ++r) {
        const int row = mBase + i * 16 + mlo + r;
        float v = acc[r] + bc;
        if (do_relu) v = fmaxf(v, 0.f);
        O[(size_t)row * D_DIM + col] = (bf16)v;
      }
    }
  }
}

// ---------------------------------------------------------------- recurrence
// One WG per (batch, 64-wide v-slice). Full f32 state slice lives in LDS.

__global__ __launch_bounds__(256, 1) void recur_kernel(
    const bf16* __restrict__ Qg, const bf16* __restrict__ Kg,
    const bf16* __restrict__ Vg, const float* __restrict__ gamma_p,
    float* __restrict__ out) {
  extern __shared__ char smem_raw[];
  float* St    = (float*)smem_raw;                 // [SLICE][ST_STRIDE], St[j][d]
  float* As32  = St + 64 * ST_STRIDE;              // [CHUNK][A_STRIDE] masked A
  float* zv    = As32 + 64 * A_STRIDE;             // [D_DIM]
  float* qz    = zv + D_DIM;                       // [CHUNK]
  float* denom = qz + 64;                          // [CHUNK]
  float* gpow  = denom + 64;                       // [CHUNK+1]
  bf16* Vt = (bf16*)(gpow + 65);                   // [SLICE][VT_STRIDE], Vt[j][t]
  bf16* Kt = Vt + 64 * VT_STRIDE;                  // [64 d][KT_STRIDE], Kt[d][t]

  const int tid = threadIdx.x, lane = tid & 31, w = tid >> 5;
  const int b  = blockIdx.x / NSLICE;
  const int j0 = (blockIdx.x % NSLICE) * SLICE;
  const bf16* Qb = Qg + (size_t)b * S_DIM * D_DIM;
  const bf16* Kb = Kg + (size_t)b * S_DIM * D_DIM;
  const bf16* Vb = Vg + (size_t)b * S_DIM * D_DIM;
  float* outb = out + (size_t)b * S_DIM * D_DIM;

  const float gamma = gamma_p[0];
  for (int i = tid; i < 65; i += 256) gpow[i] = __powf(gamma, (float)i);
  for (int i = tid; i < D_DIM; i += 256) zv[i] = 0.f;
  for (int i = tid; i < 64 * ST_STRIDE; i += 256) St[i] = 0.f;
  __syncthreads();
  const float gC = gpow[64];

  for (int c = 0; c < NCHUNK; ++c) {
    const int t0 = c * CHUNK;
    const bf16* Qc = Qb + (size_t)t0 * D_DIM;
    const bf16* Kc = Kb + (size_t)t0 * D_DIM;

    // ---- phase 1: qz[t] = q_t . z_prev ; stage Vt[j][t] (transposed V slice)
    if (tid < 64) qz[tid] = 0.f;
    __syncthreads();
    {
      const int t = tid >> 2;
      const int dseg = (tid & 3) * 256;
      const bf16* qr = Qc + (size_t)t * D_DIM + dseg;
      float s = 0.f;
      for (int d = 0; d < 256; d += 8) {
        v8bf q8 = *(const v8bf*)(qr + d);
#pragma unroll
        for (int i = 0; i < 8; ++i) s += (float)q8[i] * zv[dseg + d + i];
      }
      atomicAdd(&qz[t], s);
    }
    for (int idx = tid; idx < CHUNK * SLICE; idx += 256) {
      const int tt = idx >> 6, j = idx & 63;
      Vt[j * VT_STRIDE + tt] = Vb[(size_t)(t0 + tt) * D_DIM + j0 + j];
    }
    __syncthreads();

    // ---- phase 2: A = Q @ K^T (reduce over D), decay-mask, store f32 to LDS
    {
      const int mt = w >> 1, nt0 = (w & 1) * 2;
      const bf16* arow = Qc + (size_t)(mt * 16 + (lane & 15)) * D_DIM;
      const bf16* b0row = Kc + (size_t)(nt0 * 16 + (lane & 15)) * D_DIM;
      const bf16* b1row = b0row + 16 * D_DIM;
      v8f acc0 = zero8(), acc1 = zero8();
#pragma unroll 4
      for (int kk = 0; kk < D_DIM / 32; ++kk) {
        const int k0 = kk * 32;
        v16bf a = loadA_bf(arow, k0, lane);
        acc0 = wmma_bf(a, loadB_bf(b0row, k0, lane), acc0);
        acc1 = wmma_bf(a, loadB_bf(b1row, k0, lane), acc1);
      }
      const int nloc = lane & 15, mlo = (lane >> 4) * 8;
#pragma unroll
      for (int nt = 0; nt < 2; ++nt) {
        v8f acc = nt ? acc1 : acc0;
        const int jj = (nt0 + nt) * 16 + nloc;
#pragma unroll
        for (int r = 0; r < 8; ++r) {
          const int t = mt * 16 + mlo + r;
          const float f = (jj <= t) ? gpow[t - jj] : 0.f;
          As32[t * A_STRIDE + jj] = acc[r] * f;
        }
      }
    }
    __syncthreads();

    // ---- phase 3: denom[t] = gamma^{t+1} * qz[t] + rowsum(A[t]) + eps
    if (tid < 64) {
      float s = 0.f;
      for (int j = 0; j < 64; ++j) s += As32[tid * A_STRIDE + j];
      denom[tid] = gpow[tid + 1] * qz[tid] + s + EPS_V;
    }
    __syncthreads();

    // ---- phase 4: Y = gamma^{t+1} * (Q @ S_prev) + A @ V ; write Y/denom
    {
      const int mt = w & 3, nt0 = (w >> 2) * 2;
      const bf16* arow = Qc + (size_t)(mt * 16 + (lane & 15)) * D_DIM;
      const float* s0row = St + (size_t)(nt0 * 16 + (lane & 15)) * ST_STRIDE;
      const float* s1row = s0row + 16 * ST_STRIDE;
      v8f acc0 = zero8(), acc1 = zero8();
#pragma unroll 2
      for (int kk = 0; kk < D_DIM / 32; ++kk) {
        const int k0 = kk * 32;
        v16bf a = loadA_bf(arow, k0, lane);
        acc0 = wmma_bf(a, loadB_f32(s0row, k0, lane), acc0);
        acc1 = wmma_bf(a, loadB_f32(s1row, k0, lane), acc1);
      }
      const int nloc = lane & 15, mlo = (lane >> 4) * 8;
#pragma unroll
      for (int r = 0; r < 8; ++r) {
        const float g = gpow[mt * 16 + mlo + r + 1];
        acc0[r] *= g;
        acc1[r] *= g;
      }
      const float* aAr = As32 + (size_t)(mt * 16 + (lane & 15)) * A_STRIDE;
      const bf16* vb0 = Vt + (size_t)(nt0 * 16 + (lane & 15)) * VT_STRIDE;
      const bf16* vb1 = vb0 + 16 * VT_STRIDE;
#pragma unroll
      for (int ks = 0; ks < 2; ++ks) {
        const int k0 = ks * 32;
        v16bf a = loadA_f32(aAr, k0, lane);
        acc0 = wmma_bf(a, loadB_bf(vb0, k0, lane), acc0);
        acc1 = wmma_bf(a, loadB_bf(vb1, k0, lane), acc1);
      }
#pragma unroll
      for (int nt = 0; nt < 2; ++nt) {
        v8f acc = nt ? acc1 : acc0;
        const int jj = (nt0 + nt) * 16 + nloc;
#pragma unroll
        for (int r = 0; r < 8; ++r) {
          const int t = mt * 16 + mlo + r;
          outb[(size_t)(t0 + t) * D_DIM + j0 + jj] = acc[r] / denom[t];
        }
      }
    }
    __syncthreads();  // everyone done reading St / As32 before state update

    // ---- phase 5: S = gamma^C * S + (scaled K)^T @ V, z update; 16 d-panels
    for (int p = 0; p < 16; ++p) {
      const int d0 = p * 64;
      for (int idx = tid; idx < CHUNK * 64; idx += 256) {
        const int tt = idx >> 6, dl = idx & 63;
        const float v =
            (float)Kc[(size_t)tt * D_DIM + d0 + dl] * gpow[63 - tt];
        Kt[dl * KT_STRIDE + tt] = (bf16)v;
      }
      __syncthreads();
      if (tid < 64) {  // z[d0+tid] update (Kt rows already gamma-scaled)
        float s = 0.f;
        for (int tt = 0; tt < 64; ++tt) s += (float)Kt[tid * KT_STRIDE + tt];
        zv[d0 + tid] = gC * zv[d0 + tid] + s;
      }
      {
        const int mt = w & 3, nt0 = (w >> 2) * 2;
        const bf16* arow = Kt + (size_t)(mt * 16 + (lane & 15)) * KT_STRIDE;
        const bf16* vb0 = Vt + (size_t)(nt0 * 16 + (lane & 15)) * VT_STRIDE;
        const bf16* vb1 = vb0 + 16 * VT_STRIDE;
        v8f acc0 = zero8(), acc1 = zero8();
#pragma unroll
        for (int ks = 0; ks < 2; ++ks) {
          const int k0 = ks * 32;
          v16bf a = loadA_bf(arow, k0, lane);
          acc0 = wmma_bf(a, loadB_bf(vb0, k0, lane), acc0);
          acc1 = wmma_bf(a, loadB_bf(vb1, k0, lane), acc1);
        }
        const int nloc = lane & 15, mlo = (lane >> 4) * 8;
#pragma unroll
        for (int nt = 0; nt < 2; ++nt) {
          v8f acc = nt ? acc1 : acc0;
          const int jj = (nt0 + nt) * 16 + nloc;
#pragma unroll
          for (int r = 0; r < 8; ++r) {
            const int dl = mt * 16 + mlo + r;
            float* sp = St + (size_t)jj * ST_STRIDE + d0 + dl;
            *sp = gC * *sp + acc[r];
          }
        }
      }
      __syncthreads();
    }
  }
}

// ---------------------------------------------------------------- host launch

extern "C" void kernel_launch(void* const* d_in, const int* in_sizes, int n_in,
                              void* d_out, int out_size, void* d_ws,
                              size_t ws_size, hipStream_t stream) {
  const float* x  = (const float*)d_in[0];
  const float* Wq = (const float*)d_in[1];
  const float* bq = (const float*)d_in[2];
  const float* Wk = (const float*)d_in[3];
  const float* bk = (const float*)d_in[4];
  const float* Wv = (const float*)d_in[5];
  const float* bv = (const float*)d_in[6];
  const float* gamma_p = (const float*)d_in[7];
  float* out = (float*)d_out;

  const size_t nX = (size_t)B_DIM * S_DIM * D_DIM;  // 8M elems
  const size_t nW = (size_t)D_DIM * D_DIM;          // 1M elems
  bf16* Xbf = (bf16*)d_ws;
  bf16* Wqb = Xbf + nX;
  bf16* Wkb = Wqb + nW;
  bf16* Wvb = Wkb + nW;
  bf16* Qg  = Wvb + nW;
  bf16* Kg  = Qg + nX;
  bf16* Vg  = Kg + nX;

  cvt_kernel<<<(int)(nX / 8 / 256), 256, 0, stream>>>(x, Xbf, (int)nX);
  cvt_kernel<<<(int)(nW / 8 / 256), 256, 0, stream>>>(Wq, Wqb, (int)nW);
  cvt_kernel<<<(int)(nW / 8 / 256), 256, 0, stream>>>(Wk, Wkb, (int)nW);
  cvt_kernel<<<(int)(nW / 8 / 256), 256, 0, stream>>>(Wv, Wvb, (int)nW);

  proj_kernel<<<dim3((B_DIM * S_DIM) / 128, D_DIM / 64, 3), 256, 0, stream>>>(
      Xbf, Wqb, Wkb, Wvb, bq, bk, bv, Qg, Kg, Vg);

  const size_t smem =
      (size_t)(64 * ST_STRIDE + 64 * A_STRIDE + D_DIM + 64 + 64 + 65) * 4 +
      (size_t)(64 * VT_STRIDE + 64 * KT_STRIDE) * 2;
  (void)hipFuncSetAttribute((const void*)recur_kernel,
                            hipFuncAttributeMaxDynamicSharedMemorySize,
                            (int)smem);
  recur_kernel<<<B_DIM * NSLICE, 256, smem, stream>>>(Qg, Kg, Vg, gamma_p, out);
}
